// LLM_35029753266228
// MI455X (gfx1250) — matrix operand order repaired and checked by reading, not compiled, not run on hardware
//
#include <hip/hip_runtime.h>
#include <hip/hip_bf16.h>

// ---- model dims (compile-time) ----
#define LN   4
#define BB   2
#define TT   1024
#define DD   1024
#define HQn  16
#define HKVn 4
#define HDn  64
#define FF   4096
#define VV   32000
#define NTOK (BB * TT)   // 2048 tokens

typedef __attribute__((ext_vector_type(16))) __bf16 v16bf;
typedef __attribute__((ext_vector_type(8)))  float  v8f;
typedef unsigned int u32x4 __attribute__((ext_vector_type(4)));
typedef int          i32x8 __attribute__((ext_vector_type(8)));
typedef int          i32x4 __attribute__((ext_vector_type(4)));

struct Frag { unsigned int d[8]; };   // 8 dwords = 16 packed bf16

__device__ __forceinline__ unsigned short f2bf(float f) {
    unsigned u = __float_as_uint(f);
    unsigned r = u + 0x7FFFu + ((u >> 16) & 1u);   // round-to-nearest-even
    return (unsigned short)(r >> 16);
}
__device__ __forceinline__ unsigned pack2(float a, float b) {
    return (unsigned)f2bf(a) | ((unsigned)f2bf(b) << 16);
}
__device__ __forceinline__ v8f wmma_bf16(const Frag& a, const Frag& b, v8f c) {
    return __builtin_amdgcn_wmma_f32_16x16x32_bf16(
        false, __builtin_bit_cast(v16bf, a),
        false, __builtin_bit_cast(v16bf, b),
        (short)0, c, false, false);
}
// A-matrix (16x32 bf16) pair index for dword j: lanes0-15 hold k {0-7,16-23},
// lanes16-31 hold k {8-15,24-31}, packed 2 per dword.
__device__ __forceinline__ int a_pair(int j, int lhalf) {
    return ((j >> 2) << 3) + (j & 3) + lhalf * 4;
}

// ---- Tensor Data Mover: 2-D bf16 tile load, D# per CDNA5 ISA ch.8 ----
// tensor: rows x cols (row stride = stride0 elements), tile td1 x td0,
// all element counts in bf16 units. lds_off = byte offset in LDS.
__device__ __forceinline__ void tdm_load_2d_bf16(
    unsigned lds_off, const unsigned short* gsrc,
    unsigned tile_d0, unsigned tile_d1,
    unsigned tensor_d0, unsigned tensor_d1, unsigned stride0)
{
    unsigned long long ga = (unsigned long long)(__SIZE_TYPE__)gsrc;
    u32x4 g0;
    g0[0] = 1u;                                      // count=1, user mode
    g0[1] = lds_off;                                 // lds_addr
    g0[2] = (unsigned)(ga & 0xffffffffu);            // global_addr[31:0]
    g0[3] = (unsigned)((ga >> 32) & 0x01ffffffu)     // global_addr[56:32]
          | (2u << 30);                              // type = 2 ("image")
    i32x8 g1;
    g1[0] = (int)(1u << 16);                         // data_size=1 (2 bytes)
    g1[1] = (int)((tensor_d0 & 0xffffu) << 16);      // [63:48]=dim0[15:0]
    g1[2] = (int)((tensor_d0 >> 16) |
                  ((tensor_d1 & 0xffffu) << 16));    // dim0[31:16] | dim1[15:0]
    g1[3] = (int)((tensor_d1 >> 16) |
                  (tile_d0 << 16));                  // dim1[31:16] | tile_dim0
    g1[4] = (int)(tile_d1 & 0xffffu);                // tile_dim1 ; tile_dim2=0
    g1[5] = (int)stride0;                            // dim0_stride[31:0]
    g1[6] = 0;                                       // dim0_stride[47:32], dim1_stride lo
    g1[7] = 0;                                       // dim1_stride hi
    i32x4 z4 = {0, 0, 0, 0};
    i32x8 z8 = {0, 0, 0, 0, 0, 0, 0, 0};
#if __has_builtin(__builtin_amdgcn_tensor_load_to_lds)
    __builtin_amdgcn_tensor_load_to_lds(g0, g1, z4, z4, z8, 0);
#endif
}
__device__ __forceinline__ void tdm_wait(void) {
#if __has_builtin(__builtin_amdgcn_s_wait_tensorcnt)
    __builtin_amdgcn_s_wait_tensorcnt((short)0);
#endif
}
__device__ __forceinline__ void tdm_wait2(void) {
#if __has_builtin(__builtin_amdgcn_s_wait_tensorcnt)
    __builtin_amdgcn_s_wait_tensorcnt((short)2);
#endif
}

// ======================= TDM-fed bf16 WMMA GEMM =======================
// C[M,N] = A[M,K](bf16) * W[N,K](bf16)^T + bias[N] (+ addsrc[M,N])
// M%64==0, N%128==0, K%64==0 (all shapes in this model satisfy this)
#define BM 64
#define BN 128
#define BK 64

template <bool ADD>
__global__ __launch_bounds__(256) void gemm_bf16_kernel(
    const unsigned short* __restrict__ A, const unsigned short* __restrict__ W,
    const float* __restrict__ bias, const float* __restrict__ addsrc,
    float* __restrict__ C, int M, int N, int K)
{
    __shared__ unsigned short As[2][BM * BK];   // 2 x 8 KB  [m][k]
    __shared__ unsigned short Bs[2][BN * BK];   // 2 x 16 KB [n][k]

    const int tid  = threadIdx.x;
    const int bm   = blockIdx.y * BM;
    const int bn   = blockIdx.x * BN;
    const int wave = tid >> 5;
    const int lane = tid & 31;
    const int wm   = (wave >> 2) * 32;       // 0 / 32
    const int wn   = (wave & 3) * 32;        // 0..96
    const int l16  = lane & 15;
    const int lhf  = lane >> 4;

    v8f acc[2][2] = {};

    const int nk = K / BK;
    if (wave == 0) {   // TDM issue is wave-level; one wave drives the DMA
        tdm_load_2d_bf16((unsigned)(__SIZE_TYPE__)&As[0][0],
                         A + (size_t)bm * K, BK, BM, K, M - bm, K);
        tdm_load_2d_bf16((unsigned)(__SIZE_TYPE__)&Bs[0][0],
                         W + (size_t)bn * K, BK, BN, K, N - bn, K);
    }
    for (int i = 0; i < nk; ++i) {
        if (wave == 0) {
            if (i + 1 < nk) {
                int k1 = (i + 1) * BK;
                tdm_load_2d_bf16((unsigned)(__SIZE_TYPE__)&As[(i + 1) & 1][0],
                                 A + (size_t)bm * K + k1, BK, BM, K - k1, M - bm, K);
                tdm_load_2d_bf16((unsigned)(__SIZE_TYPE__)&Bs[(i + 1) & 1][0],
                                 W + (size_t)bn * K + k1, BK, BN, K - k1, N - bn, K);
                tdm_wait2();     // tile i landed (TENSORcnt is in-order)
            } else {
                tdm_wait();
            }
        }
        __syncthreads();

        const unsigned* Asu = (const unsigned*)As[i & 1];   // row stride 32 dwords
        const unsigned* Bsu = (const unsigned*)Bs[i & 1];
#pragma unroll
        for (int ks = 0; ks < 2; ++ks) {
            Frag af[2], bf[2];
#pragma unroll
            for (int im = 0; im < 2; ++im) {
                int m = wm + im * 16 + l16;
#pragma unroll
                for (int j = 0; j < 8; ++j)
                    af[im].d[j] = Asu[m * 32 + ks * 16 + a_pair(j, lhf)];
            }
#pragma unroll
            for (int in = 0; in < 2; ++in) {
                int n = wn + in * 16 + l16;
#pragma unroll
                for (int j = 0; j < 8; ++j)
                    bf[in].d[j] = Bsu[n * 32 + ks * 16 + a_pair(j, lhf)];
            }
#pragma unroll
            for (int im = 0; im < 2; ++im)
#pragma unroll
                for (int in = 0; in < 2; ++in)
                    acc[im][in] = wmma_bf16(af[im], bf[in], acc[im][in]);
        }
        __syncthreads();
    }

    // epilogue: C-layout VGPR r -> row r (lanes 0-15) / row 8+r (lanes 16-31)
#pragma unroll
    for (int im = 0; im < 2; ++im) {
        int mbase = bm + wm + im * 16 + lhf * 8;
#pragma unroll
        for (int in = 0; in < 2; ++in) {
            int n = bn + wn + in * 16 + l16;
            float bv = bias[n];
#pragma unroll
            for (int r = 0; r < 8; ++r) {
                size_t idx = (size_t)(mbase + r) * N + n;
                float v = acc[im][in][r] + bv;
                if (ADD) v += addsrc[idx];
                C[idx] = v;
            }
        }
    }
}

// ======================= flash attention (WMMA, all-bf16 inputs) ========
// one wave per (b, h, 16-query tile); online softmax over 32-key blocks
// Q16: (B,T,HQ,64) bf16 roped; K16: (B,T,HKV,64) bf16 roped;
// V16t: (B,HKV,64,T) bf16 (transposed so P*V B-frags are contiguous)
__global__ __launch_bounds__(128) void attn_kernel(
    const unsigned short* __restrict__ Q16,
    const unsigned short* __restrict__ K16,
    const unsigned short* __restrict__ V16t,
    unsigned short* __restrict__ O)    // (B,T,HQ*HD) bf16 ctx
{
    __shared__ unsigned PldsAll[4][16 * 16];       // per-wave 16x32 bf16
    const int wave = threadIdx.x >> 5;
    const int lane = threadIdx.x & 31;
    unsigned* Plds = PldsAll[wave];

    const int gw  = blockIdx.x * 4 + wave;
    const int nq  = TT / 16;                        // 64
    const int qt  = gw % nq;
    const int h   = (gw / nq) % HQn;
    const int b   = gw / (nq * HQn);
    const int kvh = h / (HQn / HKVn);
    const int l16 = lane & 15;
    const int lhf = lane >> 4;

    // Q fragments (A-layout): direct dword loads of packed bf16
    const int trow = qt * 16 + l16;
    const unsigned* qrow =
        (const unsigned*)(Q16 + ((size_t)(b * TT + trow)) * (HQn * HDn) + h * HDn);
    Frag qf[2];
#pragma unroll
    for (int ks = 0; ks < 2; ++ks)
#pragma unroll
        for (int j = 0; j < 8; ++j)
            qf[ks].d[j] = qrow[ks * 16 + a_pair(j, lhf)];

    float mrow[8], lrow[8];
#pragma unroll
    for (int r = 0; r < 8; ++r) { mrow[r] = -1e30f; lrow[r] = 0.0f; }
    v8f accO[4] = {};

    const int nkb = (qt * 16 + 47) / 32;            // causal key blocks
    for (int kb = 0; kb < nkb; ++kb) {
        const int key0 = kb * 32;

        // S = Q K^T  (two 16x16 tiles over 32 keys; 2 k-steps over HD=64)
        v8f s[2] = {};
#pragma unroll
        for (int nt = 0; nt < 2; ++nt) {
            int key = key0 + nt * 16 + l16;
            const unsigned* krow =
                (const unsigned*)(K16 + ((size_t)(b * TT + key)) * (HKVn * HDn) + kvh * HDn);
#pragma unroll
            for (int ks = 0; ks < 2; ++ks) {
                Frag kf;
#pragma unroll
                for (int j = 0; j < 8; ++j)
                    kf.d[j] = krow[ks * 16 + a_pair(j, lhf)];
                s[nt] = wmma_bf16(qf[ks], kf, s[nt]);
            }
        }
        // scale by 1/sqrt(HD) + causal mask (branch-free; EXEC stays all-1s)
#pragma unroll
        for (int nt = 0; nt < 2; ++nt) {
            int key = key0 + nt * 16 + l16;
#pragma unroll
            for (int r = 0; r < 8; ++r) {
                int t = qt * 16 + r + lhf * 8;
                s[nt][r] = (key > t) ? -1e30f : s[nt][r] * 0.125f;
            }
        }
        // online softmax: row stats via 16-lane xor-shuffle reductions
        float alpha[8];
#pragma unroll
        for (int r = 0; r < 8; ++r) {
            float v = fmaxf(s[0][r], s[1][r]);
            for (int off = 1; off < 16; off <<= 1)
                v = fmaxf(v, __shfl_xor(v, off, 16));
            float mnew = fmaxf(mrow[r], v);
            alpha[r] = __expf(mrow[r] - mnew);
            mrow[r] = mnew;
        }
#pragma unroll
        for (int r = 0; r < 8; ++r) {
            s[0][r] = __expf(s[0][r] - mrow[r]);
            s[1][r] = __expf(s[1][r] - mrow[r]);
            float v = s[0][r] + s[1][r];
            for (int off = 1; off < 16; off <<= 1)
                v += __shfl_xor(v, off, 16);
            lrow[r] = lrow[r] * alpha[r] + v;
        }
#pragma unroll
        for (int nt = 0; nt < 4; ++nt)
#pragma unroll
            for (int r = 0; r < 8; ++r) accO[nt][r] *= alpha[r];

        // P: C-layout -> A-layout transpose through per-wave LDS (bf16)
        unsigned short* P16 = (unsigned short*)Plds;
#pragma unroll
        for (int nt = 0; nt < 2; ++nt)
#pragma unroll
            for (int r = 0; r < 8; ++r)
                P16[(r + lhf * 8) * 32 + nt * 16 + l16] = f2bf(s[nt][r]);
        Frag pf;
#pragma unroll
        for (int j = 0; j < 8; ++j) pf.d[j] = Plds[l16 * 16 + a_pair(j, lhf)];

        // O += P V (four 16-col tiles over HD=64); V transposed -> contiguous
#pragma unroll
        for (int nt = 0; nt < 4; ++nt) {
            int hd = nt * 16 + l16;
            const unsigned* vrow =
                (const unsigned*)(V16t + (((size_t)(b * HKVn + kvh)) * HDn + hd) * TT);
            Frag vf;
#pragma unroll
            for (int j = 0; j < 8; ++j)
                vf.d[j] = vrow[(key0 >> 1) + a_pair(j, lhf)];
            accO[nt] = wmma_bf16(pf, vf, accO[nt]);
        }
    }

    // write ctx (bf16)
#pragma unroll
    for (int nt = 0; nt < 4; ++nt)
#pragma unroll
        for (int r = 0; r < 8; ++r) {
            int t = qt * 16 + r + lhf * 8;
            int col = h * HDn + nt * 16 + l16;
            O[((size_t)(b * TT + t)) * (HQn * HDn) + col] = f2bf(accO[nt][r] / lrow[r]);
        }
}

// ======================= elementwise kernels =======================
__global__ __launch_bounds__(256) void cvt_bf16_kernel(
    const float* __restrict__ src, unsigned short* __restrict__ dst, int n)
{
    int i = blockIdx.x * 256 + threadIdx.x;
    if (i < n) dst[i] = f2bf(src[i]);
}

__global__ __launch_bounds__(256) void embed_kernel(
    const int* __restrict__ tokens, const float* __restrict__ W, float* __restrict__ x)
{
    int idx = blockIdx.x * 256 + threadIdx.x;
    if (idx >= NTOK * DD) return;
    int tok = tokens[idx / DD];
    x[idx] = W[(size_t)tok * DD + (idx % DD)];
}

__global__ __launch_bounds__(256) void rmsnorm_kernel(
    const float* __restrict__ x, const float* __restrict__ w,
    unsigned short* __restrict__ out)
{
    __shared__ float red[9];
    int row = blockIdx.x;
    const float* xr = x + (size_t)row * DD;
    float s = 0.0f;
    for (int i = threadIdx.x; i < DD; i += 256) { float v = xr[i]; s += v * v; }
    for (int off = 16; off > 0; off >>= 1) s += __shfl_xor(s, off, 32);
    if ((threadIdx.x & 31) == 0) red[threadIdx.x >> 5] = s;
    __syncthreads();
    if (threadIdx.x == 0) {
        float t = 0.0f;
        for (int i = 0; i < 8; ++i) t += red[i];
        red[8] = rsqrtf(t / (float)DD + 1e-6f);
    }
    __syncthreads();
    float rinv = red[8];
    for (int i = threadIdx.x; i < DD; i += 256)
        out[(size_t)row * DD + i] = f2bf(xr[i] * rinv * w[i]);
}

// RoPE: read fp32 projection, write roped bf16 (x: (B,T,H,64))
__global__ __launch_bounds__(256) void rope_bf16_kernel(
    const float* __restrict__ x, unsigned short* __restrict__ out, int H)
{
    int idx = blockIdx.x * 256 + threadIdx.x;
    int total = NTOK * H * 32;
    if (idx >= total) return;
    int i  = idx & 31;
    int h  = (idx >> 5) % H;
    int bt = idx / (32 * H);
    int t  = bt % TT;
    size_t base = (size_t)bt * H * HDn + h * HDn;
    float inv = __powf(10000.0f, -(2.0f * (float)i) / 64.0f);
    float ang = (float)t * inv;
    float c = __cosf(ang), s = __sinf(ang);
    float x1 = x[base + i], x2 = x[base + i + 32];
    out[base + i]      = f2bf(x1 * c - x2 * s);
    out[base + i + 32] = f2bf(x2 * c + x1 * s);
}

// V transpose: (B,T,HKV,64) f32 -> (B,HKV,64,T) bf16
__global__ __launch_bounds__(256) void vtranspose_kernel(
    const float* __restrict__ v, unsigned short* __restrict__ out)
{
    int idx = blockIdx.x * 256 + threadIdx.x;
    if (idx >= NTOK * HKVn * HDn) return;
    int hd  = idx & (HDn - 1);
    int kvh = (idx / HDn) & (HKVn - 1);
    int bt  = idx / (HDn * HKVn);
    int t   = bt % TT;
    int b   = bt / TT;
    out[(((size_t)(b * HKVn + kvh)) * HDn + hd) * TT + t] = f2bf(v[idx]);
}

__global__ __launch_bounds__(256) void silumul_kernel(
    const float* __restrict__ g, const float* __restrict__ u,
    unsigned short* __restrict__ out)
{
    int idx = blockIdx.x * 256 + threadIdx.x;
    if (idx >= NTOK * FF) return;
    float gv = g[idx];
    out[idx] = f2bf((gv / (1.0f + __expf(-gv))) * u[idx]);
}

// ======================= host driver =======================
typedef unsigned short bf16_t;

static inline void launch_gemm(const bf16_t* A, const bf16_t* W, const float* bias,
                               const float* addsrc, float* C, int M, int N, int K,
                               hipStream_t s)
{
    dim3 grid(N / BN, M / BM);
    if (addsrc)
        gemm_bf16_kernel<true ><<<grid, 256, 0, s>>>(A, W, bias, addsrc, C, M, N, K);
    else
        gemm_bf16_kernel<false><<<grid, 256, 0, s>>>(A, W, bias, addsrc, C, M, N, K);
}
static inline void launch_cvt(const float* src, bf16_t* dst, int n, hipStream_t s)
{
    cvt_bf16_kernel<<<(n + 255) / 256, 256, 0, s>>>(src, dst, n);
}

extern "C" void kernel_launch(void* const* d_in, const int* in_sizes, int n_in,
                              void* d_out, int out_size, void* d_ws, size_t ws_size,
                              hipStream_t stream)
{
    const int*   tokens  = (const int*)  d_in[0];
    const float* embed_W = (const float*)d_in[1];
    const float* q_w = (const float*)d_in[2],  *q_b = (const float*)d_in[3];
    const float* k_w = (const float*)d_in[4],  *k_b = (const float*)d_in[5];
    const float* v_w = (const float*)d_in[6],  *v_b = (const float*)d_in[7];
    const float* o_w = (const float*)d_in[8],  *o_b = (const float*)d_in[9];
    const float* attn_nw = (const float*)d_in[10];
    const float* mlp_nw  = (const float*)d_in[11];
    const float* g_w = (const float*)d_in[12], *g_b = (const float*)d_in[13];
    const float* u_w = (const float*)d_in[14], *u_b = (const float*)d_in[15];
    const float* d_w = (const float*)d_in[16], *d_b = (const float*)d_in[17];
    const float* norm_w  = (const float*)d_in[18];
    const float* head_w  = (const float*)d_in[19], *head_b = (const float*)d_in[20];

    // ---- carve workspace ----
    char* p = (char*)d_ws;
    auto alloc = [&](size_t bytes) { void* r = p; p += (bytes + 255) & ~(size_t)255; return r; };
    float*  x    = (float*)alloc((size_t)NTOK * DD * 4);
    bf16_t* h16  = (bf16_t*)alloc((size_t)NTOK * DD * 2);
    float*  q    = (float*)alloc((size_t)NTOK * DD * 4);
    float*  kbuf = (float*)alloc((size_t)NTOK * HKVn * HDn * 4);
    float*  vbuf = (float*)alloc((size_t)NTOK * HKVn * HDn * 4);
    bf16_t* q16  = (bf16_t*)alloc((size_t)NTOK * DD * 2);
    bf16_t* k16  = (bf16_t*)alloc((size_t)NTOK * HKVn * HDn * 2);
    bf16_t* v16t = (bf16_t*)alloc((size_t)NTOK * HKVn * HDn * 2);
    bf16_t* ctx  = (bf16_t*)alloc((size_t)NTOK * DD * 2);
    float*  gbuf = (float*)alloc((size_t)NTOK * FF * 4);
    float*  ubuf = (float*)alloc((size_t)NTOK * FF * 4);
    bf16_t* gu16 = (bf16_t*)alloc((size_t)NTOK * FF * 2);
    const int QWN = LN * HQn * HDn * DD, KWN = LN * HKVn * HDn * DD;
    const int GWN = LN * FF * DD, HWN = VV * DD;
    bf16_t* qw16 = (bf16_t*)alloc((size_t)QWN * 2);
    bf16_t* kw16 = (bf16_t*)alloc((size_t)KWN * 2);
    bf16_t* vw16 = (bf16_t*)alloc((size_t)KWN * 2);
    bf16_t* ow16 = (bf16_t*)alloc((size_t)QWN * 2);
    bf16_t* gw16 = (bf16_t*)alloc((size_t)GWN * 2);
    bf16_t* uw16 = (bf16_t*)alloc((size_t)GWN * 2);
    bf16_t* dw16 = (bf16_t*)alloc((size_t)GWN * 2);
    bf16_t* hw16 = (bf16_t*)alloc((size_t)HWN * 2);

    // ---- one-shot weight conversion to bf16 (halves streamed bytes) ----
    launch_cvt(q_w, qw16, QWN, stream);
    launch_cvt(k_w, kw16, KWN, stream);
    launch_cvt(v_w, vw16, KWN, stream);
    launch_cvt(o_w, ow16, QWN, stream);
    launch_cvt(g_w, gw16, GWN, stream);
    launch_cvt(u_w, uw16, GWN, stream);
    launch_cvt(d_w, dw16, GWN, stream);
    launch_cvt(head_w, hw16, HWN, stream);

    // ---- embedding ----
    embed_kernel<<<(NTOK * DD + 255) / 256, 256, 0, stream>>>(tokens, embed_W, x);

    for (int l = 0; l < LN; ++l) {
        const bf16_t* qw = qw16 + (size_t)l * (HQn * HDn) * DD;
        const bf16_t* kw = kw16 + (size_t)l * (HKVn * HDn) * DD;
        const bf16_t* vw = vw16 + (size_t)l * (HKVn * HDn) * DD;
        const bf16_t* ow = ow16 + (size_t)l * DD * (HQn * HDn);
        const bf16_t* gw = gw16 + (size_t)l * FF * DD;
        const bf16_t* uw = uw16 + (size_t)l * FF * DD;
        const bf16_t* dw = dw16 + (size_t)l * DD * FF;

        rmsnorm_kernel<<<NTOK, 256, 0, stream>>>(x, attn_nw + (size_t)l * DD, h16);
        launch_gemm(h16, qw, q_b + (size_t)l * HQn * HDn,  nullptr, q,    NTOK, HQn * HDn,  DD, stream);
        launch_gemm(h16, kw, k_b + (size_t)l * HKVn * HDn, nullptr, kbuf, NTOK, HKVn * HDn, DD, stream);
        launch_gemm(h16, vw, v_b + (size_t)l * HKVn * HDn, nullptr, vbuf, NTOK, HKVn * HDn, DD, stream);
        rope_bf16_kernel<<<(NTOK * HQn * 32 + 255) / 256, 256, 0, stream>>>(q, q16, HQn);
        rope_bf16_kernel<<<(NTOK * HKVn * 32 + 255) / 256, 256, 0, stream>>>(kbuf, k16, HKVn);
        vtranspose_kernel<<<(NTOK * HKVn * HDn + 255) / 256, 256, 0, stream>>>(vbuf, v16t);
        attn_kernel<<<(BB * HQn * (TT / 16)) / 4, 128, 0, stream>>>(q16, k16, v16t, ctx);
        launch_gemm(ctx, ow, o_b + (size_t)l * DD, x, x, NTOK, DD, HQn * HDn, stream);

        rmsnorm_kernel<<<NTOK, 256, 0, stream>>>(x, mlp_nw + (size_t)l * DD, h16);
        launch_gemm(h16, gw, g_b + (size_t)l * FF, nullptr, gbuf, NTOK, FF, DD, stream);
        launch_gemm(h16, uw, u_b + (size_t)l * FF, nullptr, ubuf, NTOK, FF, DD, stream);
        silumul_kernel<<<(NTOK * FF + 255) / 256, 256, 0, stream>>>(gbuf, ubuf, gu16);
        launch_gemm(gu16, dw, d_b + (size_t)l * DD, x, x, NTOK, DD, FF, stream);
    }

    rmsnorm_kernel<<<NTOK, 256, 0, stream>>>(x, norm_w, h16);
    launch_gemm(h16, hw16, head_b, nullptr, (float*)d_out, NTOK, VV, DD, stream);

    (void)in_sizes; (void)n_in; (void)out_size; (void)ws_size;
}
